// Seq2SeqSum_29274497089879
// MI455X (gfx1250) — matrix operand order repaired and checked by reading, not compiled, not run on hardware
//
// CDNA5 (gfx1250, MI455X) seq2seq LSTM + attention.
// bf16 WMMA (v_wmma_f32_16x16x32_bf16) with fp32 accumulation for all GEMMs,
// fp32 VALU for gates/softmax/log-softmax. Recurrences run as persistent
// single-workgroup WMMA kernels with LDS-resident [x|h] state, fed by
// gfx1250 async global->LDS DMA (global_load_async_to_lds_b128 / ASYNCcnt).
#include <hip/hip_runtime.h>
#include <stdint.h>

#define S_LEN 400
#define T_LEN 64
#define B_SZ  32
#define V_SZ  32000
#define E_SZ  256
#define H_SZ  512
#define G4H   2048
#define KXH   768      // E + H (concatenated GEMM depth)
#define XH_LD 776      // padded LDS row stride (bank-conflict-free b128 reads)

typedef __bf16 bf16_t;
typedef bf16_t v16bf __attribute__((ext_vector_type(16)));
typedef float  v8f   __attribute__((ext_vector_type(8)));

union Frag16 { v16bf v; uint4 q[2]; };

__device__ __forceinline__ unsigned short f2bf(float f) {
  unsigned int x = __float_as_uint(f);
  x += 0x7fffu + ((x >> 16) & 1u);            // round-to-nearest-even
  return (unsigned short)(x >> 16);
}

__device__ __forceinline__ float sigmf(float x) { return 1.f / (1.f + __expf(-x)); }

__device__ __forceinline__ v8f wmma_bf16(v16bf a, v16bf b, v8f c) {
  return __builtin_amdgcn_wmma_f32_16x16x32_bf16(false, a, false, b, (short)0, c,
                                                 false, false);
}

// Async DMA: copy 16B/lane global -> LDS (GVS mode: SGPR base + 32-bit VGPR
// byte offset; VDST VGPR holds LDS byte address). Tracked by ASYNCcnt.
__device__ __forceinline__ void async_g2l_b128(const void* sbase,
                                               unsigned goff_bytes,
                                               unsigned lds_bytes) {
  asm volatile("global_load_async_to_lds_b128 %0, %1, %2"
               :: "v"(lds_bytes), "v"(goff_bytes), "s"(sbase)
               : "memory");
}
__device__ __forceinline__ void wait_async0() {
  asm volatile("s_wait_asynccnt 0" ::: "memory");
}

// A fragment (16x32 bf16, M x K) from row-major A[m][k].
// lane<16: M=lane, K = k0+{0..7} (v0-3), k0+{16..23} (v4-7)
// lane>=16: M=lane-16, K = k0+{8..15}, k0+{24..31}
__device__ __forceinline__ v16bf load_a_frag(const unsigned short* A, int lda,
                                             int m0, int k0, int lane) {
  int m  = m0 + (lane & 15);
  int kb = k0 + ((lane >> 4) << 3);
  Frag16 f;
  f.q[0] = *(const uint4*)(A + (long)m * lda + kb);
  f.q[1] = *(const uint4*)(A + (long)m * lda + kb + 16);
  return f.v;
}

// B fragment (32x16 bf16, K x N) built from row-major W[n][k] (i.e. B = W^T).
// lane<16: N=lane, K=k0+{0..7},{16..23}; lane>=16: N=lane-16, K=k0+{8..15},{24..31}.
__device__ __forceinline__ v16bf load_b_frag(const unsigned short* W, int ldw,
                                             int n0, int k0, int lane) {
  int n  = n0 + (lane & 15);
  int kb = k0 + ((lane >> 4) << 3);
  Frag16 f;
  f.q[0] = *(const uint4*)(W + (long)n * ldw + kb);
  f.q[1] = *(const uint4*)(W + (long)n * ldw + kb + 16);
  return f.v;
}

// ---------------------------------------------------------------- prep kernels
__global__ void f32_to_bf16_kernel(const float* __restrict__ in,
                                   unsigned short* __restrict__ out, int n) {
  int i = blockIdx.x * 256 + threadIdx.x;
  if (i < n) out[i] = f2bf(in[i]);
}

// Wcat[n][0:256]=W_ih[n], Wcat[n][256:768]=W_hh[n]; bcat = b_ih + b_hh
__global__ void build_wcat_kernel(const float* __restrict__ Wih,
                                  const float* __restrict__ Whh,
                                  const float* __restrict__ bih,
                                  const float* __restrict__ bhh,
                                  unsigned short* __restrict__ Wcat,
                                  float* __restrict__ bcat) {
  int idx = blockIdx.x * 256 + threadIdx.x;
  if (idx >= G4H * KXH) return;
  int n = idx / KXH, k = idx - n * KXH;
  float v = (k < E_SZ) ? Wih[n * E_SZ + k] : Whh[n * H_SZ + (k - E_SZ)];
  Wcat[idx] = f2bf(v);
  if (k == 0) bcat[n] = bih[n] + bhh[n];
}

// xe[s][b][e] = emb_bf[src[b][s]][e]
__global__ void embed_src_kernel(const unsigned short* __restrict__ emb_bf,
                                 const int* __restrict__ src,
                                 unsigned short* __restrict__ xe) {
  int idx = blockIdx.x * 256 + threadIdx.x;
  if (idx >= S_LEN * B_SZ * E_SZ) return;
  int e = idx & (E_SZ - 1);
  int sb = idx >> 8;
  int b = sb & (B_SZ - 1);
  int s = sb >> 5;
  int id = src[b * S_LEN + s];
  xe[idx] = emb_bf[(long)id * E_SZ + e];
}

// ------------------------------------------------- persistent bi-LSTM encoder
// grid = 2 blocks (dir 0 fwd / 1 bwd), 512 threads = 16 waves.
// Wave w owns gate columns j in [32w, 32w+32) for all 4 gates -> gate fusion
// is lane-local. The two 16-wide j-subtiles are processed SEQUENTIALLY so only
// 64 accumulator VGPRs are live at once (no spills); new h is staged in
// registers and committed to LDS after the post-WMMA barrier.
__global__ __launch_bounds__(512) void enc_birnn_kernel(
    const unsigned short* __restrict__ xe,
    const unsigned short* __restrict__ Wcat_f,
    const unsigned short* __restrict__ Wcat_b,
    const float* __restrict__ bcat_f, const float* __restrict__ bcat_b,
    const float* __restrict__ init_h, const float* __restrict__ init_c,
    float* __restrict__ enc_outs,            // (S,B,1024) f32
    unsigned short* __restrict__ enc_outs_bf,
    unsigned short* __restrict__ hcat_bf,    // (B,1024) final h
    unsigned short* __restrict__ ccat_bf) {  // (B,1024) final c
  __shared__ unsigned short xh[B_SZ * XH_LD];   // LDS offset 0
  const int dir = blockIdx.x;
  const unsigned short* Wcat = dir ? Wcat_b : Wcat_f;
  const float* bcat = dir ? bcat_b : bcat_f;
  const int tid = threadIdx.x, lane = tid & 31, w = tid >> 5;
  const int lane15 = lane & 15, hi = lane >> 4;
  const int j0 = w * 32;

  float breg[2][4];
  float creg[32];
#pragma unroll
  for (int jt = 0; jt < 2; ++jt) {
    int j = j0 + jt * 16 + lane15;
#pragma unroll
    for (int g = 0; g < 4; ++g) breg[jt][g] = bcat[g * H_SZ + j];
  }
#pragma unroll
  for (int mt = 0; mt < 2; ++mt)
#pragma unroll
    for (int jt = 0; jt < 2; ++jt) {
      int j = j0 + jt * 16 + lane15;
      float c0 = init_c[dir * H_SZ + j];
      unsigned short h0 = f2bf(init_h[dir * H_SZ + j]);
#pragma unroll
      for (int r = 0; r < 8; ++r) {
        creg[(mt * 2 + jt) * 8 + r] = c0;
        int m = mt * 16 + r + 8 * hi;
        xh[m * XH_LD + E_SZ + j] = h0;
      }
    }
  {  // async DMA the t=0 x tile into LDS
    int sidx = dir ? (S_LEN - 1) : 0;
    for (int c = tid; c < B_SZ * E_SZ / 8; c += 512) {
      int m = c >> 5, e0 = (c & 31) * 8;
      unsigned lds = (unsigned)((m * XH_LD + e0) * 2);
      unsigned gof = (unsigned)((((sidx * B_SZ + m) * E_SZ) + e0) * 2);
      async_g2l_b128(xe, gof, lds);
    }
    wait_async0();
  }
  __syncthreads();

  for (int t = 0; t < S_LEN; ++t) {
    const int sidx = dir ? (S_LEN - 1 - t) : t;
    unsigned int hstash[2][2][4];   // [jt][mt][r/2], packed bf16 pairs
    const v8f vz = {0.f, 0.f, 0.f, 0.f, 0.f, 0.f, 0.f, 0.f};

#pragma unroll 1
    for (int jt = 0; jt < 2; ++jt) {
      v8f acc[2][4];
#pragma unroll
      for (int mt = 0; mt < 2; ++mt)
#pragma unroll
        for (int g = 0; g < 4; ++g) acc[mt][g] = vz;

      for (int kt = 0; kt < KXH / 32; ++kt) {
        int k0 = kt * 32;
        v16bf a0 = load_a_frag(xh, XH_LD, 0, k0, lane);
        v16bf a1 = load_a_frag(xh, XH_LD, 16, k0, lane);
#pragma unroll
        for (int g = 0; g < 4; ++g) {
          v16bf bb = load_b_frag(Wcat, KXH, g * H_SZ + j0 + jt * 16, k0, lane);
          acc[0][g] = wmma_bf16(a0, bb, acc[0][g]);
          acc[1][g] = wmma_bf16(a1, bb, acc[1][g]);
        }
      }
      const int j = j0 + jt * 16 + lane15;
#pragma unroll
      for (int mt = 0; mt < 2; ++mt) {
#pragma unroll
        for (int r = 0; r < 8; ++r) {
          int m = mt * 16 + r + 8 * hi;
          float ip = acc[mt][0][r] + breg[jt][0];
          float fp = acc[mt][1][r] + breg[jt][1];
          float gp = acc[mt][2][r] + breg[jt][2];
          float op = acc[mt][3][r] + breg[jt][3];
          int ci = (mt * 2 + jt) * 8 + r;
          float cv = sigmf(fp) * creg[ci] + sigmf(ip) * tanhf(gp);
          creg[ci] = cv;
          float hv = sigmf(op) * tanhf(cv);
          unsigned short hb = f2bf(hv);
          if (r & 1) hstash[jt][mt][r >> 1] |= ((unsigned)hb) << 16;
          else       hstash[jt][mt][r >> 1] = (unsigned)hb;
          long go = ((long)sidx * B_SZ + m) * 1024 + dir * H_SZ + j;
          enc_outs[go] = hv;
          enc_outs_bf[go] = hb;
          if (t == S_LEN - 1) {
            hcat_bf[m * 1024 + dir * H_SZ + j] = hb;
            ccat_bf[m * 1024 + dir * H_SZ + j] = f2bf(cv);
          }
        }
      }
    }
    __syncthreads();  // all LDS reads of this step complete

    // commit staged h to LDS for the next step's A operand
#pragma unroll
    for (int jt = 0; jt < 2; ++jt) {
      int j = j0 + jt * 16 + lane15;
#pragma unroll
      for (int mt = 0; mt < 2; ++mt)
#pragma unroll
        for (int i = 0; i < 4; ++i) {
          unsigned v = hstash[jt][mt][i];
          int m = mt * 16 + 2 * i + 8 * hi;
          xh[m * XH_LD + E_SZ + j] = (unsigned short)(v & 0xffffu);
          xh[(m + 1) * XH_LD + E_SZ + j] = (unsigned short)(v >> 16);
        }
    }
    // async DMA next x tile
    if (t + 1 < S_LEN) {
      int s2 = dir ? (S_LEN - 2 - t) : (t + 1);
      for (int c = tid; c < B_SZ * E_SZ / 8; c += 512) {
        int m = c >> 5, e0 = (c & 31) * 8;
        unsigned lds = (unsigned)((m * XH_LD + e0) * 2);
        unsigned gof = (unsigned)((((s2 * B_SZ + m) * E_SZ) + e0) * 2);
        async_g2l_b128(xe, gof, lds);
      }
      wait_async0();
    }
    __syncthreads();
  }
}

// ----------------------------------------------------- decoder LSTM cell step
__global__ __launch_bounds__(512) void dec_cell_kernel(
    const unsigned short* __restrict__ emb_bf,
    const int* __restrict__ tgt, int t,
    const unsigned short* __restrict__ Wcat,
    const float* __restrict__ bcat,
    const unsigned short* h_in_bf,   // (B,512) bf16, may alias h_bf
    float* cbuf,                     // (B,512) f32 in/out
    float* __restrict__ h_f32,       // (B,512)
    unsigned short* h_bf,            // (B,512)
    unsigned short* __restrict__ ctx_h) {  // (B,1536), cols 1024.. = h
  __shared__ unsigned short xh[B_SZ * XH_LD];   // LDS offset 0
  const int tid = threadIdx.x, lane = tid & 31, w = tid >> 5;
  const int lane15 = lane & 15, hi = lane >> 4;
  const int j0 = w * 32;

  // async DMA: embedded x rows (gathered via per-lane offsets) and h_prev
  for (int c = tid; c < B_SZ * E_SZ / 8; c += 512) {
    int m = c >> 5, e0 = (c & 31) * 8;
    int id = tgt[m * T_LEN + t];
    unsigned lds = (unsigned)((m * XH_LD + e0) * 2);
    unsigned gof = (unsigned)((id * E_SZ + e0) * 2);
    async_g2l_b128(emb_bf, gof, lds);
  }
  for (int c = tid; c < B_SZ * H_SZ / 8; c += 512) {
    int m = c >> 6, e0 = (c & 63) * 8;
    unsigned lds = (unsigned)((m * XH_LD + E_SZ + e0) * 2);
    unsigned gof = (unsigned)((m * H_SZ + e0) * 2);
    async_g2l_b128(h_in_bf, gof, lds);
  }
  wait_async0();
  __syncthreads();

  float breg[2][4];
#pragma unroll
  for (int jt = 0; jt < 2; ++jt) {
    int j = j0 + jt * 16 + lane15;
#pragma unroll
    for (int g = 0; g < 4; ++g) breg[jt][g] = bcat[g * H_SZ + j];
  }
  const v8f vz = {0.f, 0.f, 0.f, 0.f, 0.f, 0.f, 0.f, 0.f};

#pragma unroll 1
  for (int jt = 0; jt < 2; ++jt) {
    v8f acc[2][4];
#pragma unroll
    for (int mt = 0; mt < 2; ++mt)
#pragma unroll
      for (int g = 0; g < 4; ++g) acc[mt][g] = vz;

    for (int kt = 0; kt < KXH / 32; ++kt) {
      int k0 = kt * 32;
      v16bf a0 = load_a_frag(xh, XH_LD, 0, k0, lane);
      v16bf a1 = load_a_frag(xh, XH_LD, 16, k0, lane);
#pragma unroll
      for (int g = 0; g < 4; ++g) {
        v16bf bb = load_b_frag(Wcat, KXH, g * H_SZ + j0 + jt * 16, k0, lane);
        acc[0][g] = wmma_bf16(a0, bb, acc[0][g]);
        acc[1][g] = wmma_bf16(a1, bb, acc[1][g]);
      }
    }
    const int j = j0 + jt * 16 + lane15;
#pragma unroll
    for (int mt = 0; mt < 2; ++mt) {
#pragma unroll
      for (int r = 0; r < 8; ++r) {
        int m = mt * 16 + r + 8 * hi;
        float ip = acc[mt][0][r] + breg[jt][0];
        float fp = acc[mt][1][r] + breg[jt][1];
        float gp = acc[mt][2][r] + breg[jt][2];
        float op = acc[mt][3][r] + breg[jt][3];
        float cv = sigmf(fp) * cbuf[m * H_SZ + j] + sigmf(ip) * tanhf(gp);
        cbuf[m * H_SZ + j] = cv;
        float hv = sigmf(op) * tanhf(cv);
        unsigned short hb = f2bf(hv);
        h_f32[m * H_SZ + j] = hv;
        h_bf[m * H_SZ + j] = hb;
        ctx_h[m * 1536 + 1024 + j] = hb;
      }
    }
  }
}

// ------------------------------------------------------ generic bf16 WMMA GEMM
// C[M,N] = act(A[M,K] @ W[N,K]^T + bias). Block = 8 waves: 32(M) x 256(N) tile,
// each wave a 16x64 strip. M,K mult of 32; N mult of 256 (all true here).
__global__ __launch_bounds__(256) void gemm_bf16_kernel(
    const unsigned short* __restrict__ A, int lda,
    const unsigned short* __restrict__ W, int ldw,
    const float* __restrict__ bias,
    float* __restrict__ Cf, unsigned short* __restrict__ Cb, int ldc,
    int K, int act) {
  const int tid = threadIdx.x, lane = tid & 31, w = tid >> 5;
  const int m0 = blockIdx.x * 32 + (w & 1) * 16;
  const int n0 = blockIdx.y * 256 + (w >> 1) * 64;
  const v8f vz = {0.f, 0.f, 0.f, 0.f, 0.f, 0.f, 0.f, 0.f};
  v8f acc[4] = {vz, vz, vz, vz};
  for (int k0 = 0; k0 < K; k0 += 32) {
    v16bf a = load_a_frag(A, lda, m0, k0, lane);
#pragma unroll
    for (int jt = 0; jt < 4; ++jt) {
      v16bf bb = load_b_frag(W, ldw, n0 + jt * 16, k0, lane);
      acc[jt] = wmma_bf16(a, bb, acc[jt]);
    }
  }
  const int lane15 = lane & 15, hi = lane >> 4;
#pragma unroll
  for (int jt = 0; jt < 4; ++jt) {
    int n = n0 + jt * 16 + lane15;
    float bv = bias ? bias[n] : 0.f;
#pragma unroll
    for (int r = 0; r < 8; ++r) {
      int m = m0 + r + 8 * hi;
      float v = acc[jt][r] + bv;
      if (act == 1) v = tanhf(v);
      if (Cf) Cf[(long)m * ldc + n] = v;
      if (Cb) Cb[(long)m * ldc + n] = f2bf(v);
    }
  }
}

// ------------------------------- attention: scores, masked softmax, context
__global__ __launch_bounds__(256) void attn_ctx_kernel(
    const float* __restrict__ h,          // (B,512)
    const float* __restrict__ attn_keys,  // (S,B,512)
    const float* __restrict__ enc_outs,   // (S,B,1024)
    const int* __restrict__ lengths,
    unsigned short* __restrict__ ctx_h) { // (B,1536), cols 0..1023 = context
  const int b = blockIdx.x, tid = threadIdx.x;
  __shared__ float sc[S_LEN];
  __shared__ float red[256];
  __shared__ float hsh[H_SZ];
  for (int k = tid; k < H_SZ; k += 256) hsh[k] = h[b * H_SZ + k];
  __syncthreads();
  const int len = lengths[b];
  for (int s = tid; s < S_LEN; s += 256) {
    const float* kp = attn_keys + ((long)s * B_SZ + b) * H_SZ;
    float a = 0.f;
    for (int k = 0; k < H_SZ; ++k) a += hsh[k] * kp[k];
    sc[s] = (s < len) ? a : -1e18f;
  }
  __syncthreads();
  float mx = -3e38f;
  for (int s = tid; s < S_LEN; s += 256) mx = fmaxf(mx, sc[s]);
  red[tid] = mx;
  __syncthreads();
  for (int off = 128; off > 0; off >>= 1) {
    if (tid < off) red[tid] = fmaxf(red[tid], red[tid + off]);
    __syncthreads();
  }
  mx = red[0];
  __syncthreads();
  float sm = 0.f;
  for (int s = tid; s < S_LEN; s += 256) {
    float e = __expf(sc[s] - mx);
    sc[s] = e;
    sm += e;
  }
  red[tid] = sm;
  __syncthreads();
  for (int off = 128; off > 0; off >>= 1) {
    if (tid < off) red[tid] += red[tid + off];
    __syncthreads();
  }
  const float inv = 1.f / red[0];
  __syncthreads();
  for (int d = tid; d < 1024; d += 256) {
    float a = 0.f;
    for (int s = 0; s < S_LEN; ++s)
      a += sc[s] * enc_outs[((long)s * B_SZ + b) * 1024 + d];
    ctx_h[b * 1536 + d] = f2bf(a * inv);
  }
}

// ------------------------------------------------------------- log-softmax
__global__ __launch_bounds__(1024) void logsoftmax_kernel(
    const float* __restrict__ logits,  // (B,V)
    float* __restrict__ out, int t) {  // out (B,T,V)
  const int b = blockIdx.x, tid = threadIdx.x;
  __shared__ float red[1024];
  const float* row = logits + (long)b * V_SZ;
  float mx = -3e38f;
  for (int v = tid; v < V_SZ; v += 1024) mx = fmaxf(mx, row[v]);
  red[tid] = mx;
  __syncthreads();
  for (int off = 512; off > 0; off >>= 1) {
    if (tid < off) red[tid] = fmaxf(red[tid], red[tid + off]);
    __syncthreads();
  }
  mx = red[0];
  __syncthreads();
  float sm = 0.f;
  for (int v = tid; v < V_SZ; v += 1024) sm += __expf(row[v] - mx);
  red[tid] = sm;
  __syncthreads();
  for (int off = 512; off > 0; off >>= 1) {
    if (tid < off) red[tid] += red[tid + off];
    __syncthreads();
  }
  const float lse = mx + __logf(red[0]);
  float* op = out + ((long)b * T_LEN + t) * V_SZ;
  for (int v = tid; v < V_SZ; v += 1024) op[v] = row[v] - lse;
}

// ------------------------------------------------------------------ launcher
extern "C" void kernel_launch(void* const* d_in, const int* in_sizes, int n_in,
                              void* d_out, int out_size, void* d_ws,
                              size_t ws_size, hipStream_t stream) {
  (void)in_sizes; (void)n_in; (void)out_size; (void)ws_size;
  const int*   src      = (const int*)d_in[0];
  const int*   src_len  = (const int*)d_in[1];
  const int*   tgt      = (const int*)d_in[2];
  const float* emb      = (const float*)d_in[3];
  const float* eWih_f   = (const float*)d_in[4];
  const float* eWhh_f   = (const float*)d_in[5];
  const float* ebih_f   = (const float*)d_in[6];
  const float* ebhh_f   = (const float*)d_in[7];
  const float* eWih_b   = (const float*)d_in[8];
  const float* eWhh_b   = (const float*)d_in[9];
  const float* ebih_b   = (const float*)d_in[10];
  const float* ebhh_b   = (const float*)d_in[11];
  const float* init_h   = (const float*)d_in[12];
  const float* init_c   = (const float*)d_in[13];
  const float* dec_h_W  = (const float*)d_in[14];
  const float* dec_c_W  = (const float*)d_in[15];
  const float* cWih     = (const float*)d_in[16];
  const float* cWhh     = (const float*)d_in[17];
  const float* cbih     = (const float*)d_in[18];
  const float* cbhh     = (const float*)d_in[19];
  const float* attn_W   = (const float*)d_in[20];
  const float* attn_b   = (const float*)d_in[21];
  const float* concat_W = (const float*)d_in[22];
  const float* concat_b = (const float*)d_in[23];
  const float* out_W    = (const float*)d_in[24];
  const float* out_b    = (const float*)d_in[25];

  char* p = (char*)d_ws;
  auto alloc = [&](size_t bytes) -> void* {
    void* r = (void*)p;
    p += (bytes + 255) & ~(size_t)255;
    return r;
  };
  unsigned short* emb_bf  = (unsigned short*)alloc((size_t)V_SZ * E_SZ * 2);
  unsigned short* xe_bf   = (unsigned short*)alloc((size_t)S_LEN * B_SZ * E_SZ * 2);
  unsigned short* Wcat_f  = (unsigned short*)alloc((size_t)G4H * KXH * 2);
  unsigned short* Wcat_b  = (unsigned short*)alloc((size_t)G4H * KXH * 2);
  unsigned short* Wcat_c  = (unsigned short*)alloc((size_t)G4H * KXH * 2);
  float* bcat_f           = (float*)alloc(G4H * 4);
  float* bcat_b           = (float*)alloc(G4H * 4);
  float* bcat_c           = (float*)alloc(G4H * 4);
  unsigned short* attnW_bf = (unsigned short*)alloc((size_t)H_SZ * 1024 * 2);
  unsigned short* dechW_bf = (unsigned short*)alloc((size_t)H_SZ * 1024 * 2);
  unsigned short* deccW_bf = (unsigned short*)alloc((size_t)H_SZ * 1024 * 2);
  unsigned short* catW_bf  = (unsigned short*)alloc((size_t)H_SZ * 1536 * 2);
  unsigned short* outW_bf  = (unsigned short*)alloc((size_t)V_SZ * H_SZ * 2);
  float* enc_outs          = (float*)alloc((size_t)S_LEN * B_SZ * 1024 * 4);
  unsigned short* enc_outs_bf = (unsigned short*)alloc((size_t)S_LEN * B_SZ * 1024 * 2);
  float* attn_keys         = (float*)alloc((size_t)S_LEN * B_SZ * H_SZ * 4);
  unsigned short* hcat_bf  = (unsigned short*)alloc(B_SZ * 1024 * 2);
  unsigned short* ccat_bf  = (unsigned short*)alloc(B_SZ * 1024 * 2);
  float* hbuf              = (float*)alloc(B_SZ * H_SZ * 4);
  unsigned short* hbuf_bf  = (unsigned short*)alloc(B_SZ * H_SZ * 2);
  float* cbuf              = (float*)alloc(B_SZ * H_SZ * 4);
  unsigned short* ctx_h    = (unsigned short*)alloc(B_SZ * 1536 * 2);
  unsigned short* cc_bf    = (unsigned short*)alloc(B_SZ * H_SZ * 2);
  float* logits            = (float*)alloc((size_t)B_SZ * V_SZ * 4);

  auto cvt = [&](const float* in, unsigned short* out, long n) {
    f32_to_bf16_kernel<<<(unsigned)((n + 255) / 256), 256, 0, stream>>>(in, out, (int)n);
  };
  cvt(emb, emb_bf, (long)V_SZ * E_SZ);
  cvt(attn_W, attnW_bf, (long)H_SZ * 1024);
  cvt(dec_h_W, dechW_bf, (long)H_SZ * 1024);
  cvt(dec_c_W, deccW_bf, (long)H_SZ * 1024);
  cvt(concat_W, catW_bf, (long)H_SZ * 1536);
  cvt(out_W, outW_bf, (long)V_SZ * H_SZ);
  const unsigned wcg = (G4H * KXH + 255) / 256;
  build_wcat_kernel<<<wcg, 256, 0, stream>>>(eWih_f, eWhh_f, ebih_f, ebhh_f, Wcat_f, bcat_f);
  build_wcat_kernel<<<wcg, 256, 0, stream>>>(eWih_b, eWhh_b, ebih_b, ebhh_b, Wcat_b, bcat_b);
  build_wcat_kernel<<<wcg, 256, 0, stream>>>(cWih, cWhh, cbih, cbhh, Wcat_c, bcat_c);
  embed_src_kernel<<<(S_LEN * B_SZ * E_SZ + 255) / 256, 256, 0, stream>>>(emb_bf, src, xe_bf);

  enc_birnn_kernel<<<2, 512, 0, stream>>>(xe_bf, Wcat_f, Wcat_b, bcat_f, bcat_b,
                                          init_h, init_c, enc_outs, enc_outs_bf,
                                          hcat_bf, ccat_bf);

  // attn_keys = enc_outs @ attn_W^T + attn_b    (M=12800, N=512, K=1024)
  gemm_bf16_kernel<<<dim3(S_LEN * B_SZ / 32, 2), 256, 0, stream>>>(
      enc_outs_bf, 1024, attnW_bf, 1024, attn_b, attn_keys, nullptr, H_SZ, 1024, 0);
  // decoder init: dec_h = hcat @ dec_h_W^T ; dec_c = ccat @ dec_c_W^T
  gemm_bf16_kernel<<<dim3(1, 2), 256, 0, stream>>>(
      hcat_bf, 1024, dechW_bf, 1024, nullptr, hbuf, hbuf_bf, H_SZ, 1024, 0);
  gemm_bf16_kernel<<<dim3(1, 2), 256, 0, stream>>>(
      ccat_bf, 1024, deccW_bf, 1024, nullptr, cbuf, nullptr, H_SZ, 1024, 0);

  for (int t = 0; t < T_LEN; ++t) {
    dec_cell_kernel<<<1, 512, 0, stream>>>(emb_bf, tgt, t, Wcat_c, bcat_c,
                                           hbuf_bf, cbuf, hbuf, hbuf_bf, ctx_h);
    attn_ctx_kernel<<<B_SZ, 256, 0, stream>>>(hbuf, attn_keys, enc_outs, src_len, ctx_h);
    // cc = tanh([context,h] @ concat_W^T + b)   (M=32, N=512, K=1536)
    gemm_bf16_kernel<<<dim3(1, 2), 256, 0, stream>>>(
        ctx_h, 1536, catW_bf, 1536, concat_b, nullptr, cc_bf, H_SZ, 1536, 1);
    // logits = cc @ out_W^T + out_b             (M=32, N=32000, K=512)
    gemm_bf16_kernel<<<dim3(1, V_SZ / 256), 256, 0, stream>>>(
        cc_bf, 512, outW_bf, 512, out_b, logits, nullptr, V_SZ, 512, 0);
    logsoftmax_kernel<<<B_SZ, 1024, 0, stream>>>(logits, (float*)d_out, t);
  }
}
